// GINPerformancePrediction_63178968924202
// MI455X (gfx1250) — matrix operand order repaired and checked by reading, not compiled, not run on hardware
//
#include <hip/hip_runtime.h>
#include <hip/hip_bf16.h>

typedef __attribute__((ext_vector_type(16))) _Float16 v16h;
typedef __attribute__((ext_vector_type(8)))  _Float16 v8h;
typedef __attribute__((ext_vector_type(8)))  float    v8f;

#define WMMA_F16(a, b, c) \
    __builtin_amdgcn_wmma_f32_16x16x32_f16(false, (a), false, (b), (short)0, (c), false, false)

__device__ __forceinline__ float elu1(float v) {
    return v > 0.0f ? v : (__expf(v) - 1.0f);
}

// ---------------------------------------------------------------------------
// Embedding gather: h[node][0:64] = emb[x[node]][0:64]   (float4 per thread)
// ---------------------------------------------------------------------------
__global__ void gather_embed(const int* __restrict__ x, const float* __restrict__ emb,
                             float* __restrict__ h, int n) {
    int idx = blockIdx.x * 256 + threadIdx.x;
    int node = idx >> 4;
    if (node >= n) return;
    int c = (idx & 15) * 4;
    int tok = x[node];
    *(float4*)(h + (size_t)node * 64 + c) =
        *(const float4*)(emb + (size_t)tok * 64 + c);
}

// ---------------------------------------------------------------------------
// Zero helpers
// ---------------------------------------------------------------------------
__global__ void zero_f4(float4* __restrict__ p, long n4) {
    long i = (long)blockIdx.x * 256 + threadIdx.x;
    if (i < n4) p[i] = make_float4(0.f, 0.f, 0.f, 0.f);
}
__global__ void zero_f1(float* __restrict__ p, long n) {
    long i = (long)blockIdx.x * 256 + threadIdx.x;
    if (i < n) p[i] = 0.0f;
}

// ---------------------------------------------------------------------------
// Edge scatter: agg[dst[e]] += h[src[e]]   (16 threads per edge, float4 each)
// L2-resident (h = 25.6MB << 192MB L2), so atomics mostly resolve on-chip.
// ---------------------------------------------------------------------------
__global__ void scatter_edges(const float* __restrict__ h, const int* __restrict__ src,
                              const int* __restrict__ dst, float* __restrict__ agg, int e) {
    int idx = blockIdx.x * 256 + threadIdx.x;
    int eid = idx >> 4;
    if (eid >= e) return;
    int c = (idx & 15) * 4;
    int s = src[eid];
    int d = dst[eid];
    float4 v = *(const float4*)(h + (size_t)s * 64 + c);
    float* p = agg + (size_t)d * 64 + c;
    atomicAdd(p + 0, v.x);
    atomicAdd(p + 1, v.y);
    atomicAdd(p + 2, v.z);
    atomicAdd(p + 3, v.w);
}

// ---------------------------------------------------------------------------
// A-fragment assembly for V_WMMA_F32_16X16X32_F16 from a row-major 16x64 f16
// LDS tile. Per ISA 7.12.2 (16-bit A 16x32):
//   lanes 0-15 : M=lane,    halves 0-7 -> K=kb+0..7,  halves 8-15 -> K=kb+16..23
//   lanes 16-31: M=lane-16, halves 0-7 -> K=kb+8..15, halves 8-15 -> K=kb+24..31
// ---------------------------------------------------------------------------
__device__ __forceinline__ v16h load_a_frag(const _Float16* __restrict__ tile,
                                            int lane, int kb) {
    int m  = lane & 15;
    int kh = kb + ((lane < 16) ? 0 : 8);
    const _Float16* p = tile + m * 64 + kh;
    v8h lo = *(const v8h*)(p);
    v8h hi = *(const v8h*)(p + 16);
    v16h a;
#pragma unroll
    for (int j = 0; j < 8; ++j) { a[j] = lo[j]; a[8 + j] = hi[j]; }
    return a;
}

// ---------------------------------------------------------------------------
// Fused GIN MLP:  out = elu( elu((h+agg)@W1 + b1)*scale + be ) @ W2 + b2 )
// (with the layer's outer elu folded in). 128 threads = 4 waves; each wave
// owns a 16-node tile; block covers 64 nodes. Both 64x64 GEMMs run on
// v_wmma_f32_16x16x32_f16 (2 K-tiles x 4 N-tiles each).
// ---------------------------------------------------------------------------
__global__ __launch_bounds__(128) void gin_mlp(
    const float* __restrict__ h, const float* __restrict__ agg,
    const float* __restrict__ W1, const float* __restrict__ b1,
    const float* __restrict__ g,  const float* __restrict__ be,
    const float* __restrict__ W2, const float* __restrict__ b2,
    float* __restrict__ hout, int n) {

    // B fragments, swizzled so a wave reads one fragment contiguously:
    // frag f = kt*4+nt ; element (f, lane, half j) = W[K][Ncol],
    //   K = kt*32 + (lane<16 ? 0 : 8) + (j&7) + ((j>>3)<<4),  Ncol = nt*16 + (lane&15)
    __shared__ __align__(32) _Float16 lds_w1[8 * 512];
    __shared__ __align__(32) _Float16 lds_w2[8 * 512];
    __shared__ __align__(32) _Float16 lds_x[4][16 * 64];   // per-wave activation tile
    __shared__ float s_b1[64], s_scale[64], s_be[64], s_b2[64];

    const int tid  = threadIdx.x;
    const int lane = tid & 31;
    const int wave = tid >> 5;

    for (int idx = tid; idx < 4096; idx += 128) {
        int f = idx >> 9;
        int rem = idx & 511;
        int l = rem >> 4;
        int j = rem & 15;
        int kt = f >> 2, nt = f & 3;
        int K    = kt * 32 + ((l < 16) ? 0 : 8) + (j & 7) + ((j >> 3) << 4);
        int Ncol = nt * 16 + (l & 15);
        lds_w1[idx] = (_Float16)W1[K * 64 + Ncol];
        lds_w2[idx] = (_Float16)W2[K * 64 + Ncol];
    }
    if (tid < 64) {
        s_b1[tid]    = b1[tid];
        s_scale[tid] = g[tid] * rsqrtf(1.0f + 1e-5f);
        s_be[tid]    = be[tid];
        s_b2[tid]    = b2[tid];
    }
    __syncthreads();

    const int base = blockIdx.x * 64 + wave * 16;

    // Stage 0: v = h + agg  ->  f16 row-major LDS tile (each lane: half a row)
    {
        int r  = lane >> 1;
        int cb = (lane & 1) * 32;
        int node = base + r;
        if (node >= n) node = n - 1;            // clamp (store is guarded later)
        const float4* ph = (const float4*)(h   + (size_t)node * 64 + cb);
        const float4* pa = (const float4*)(agg + (size_t)node * 64 + cb);
        _Float16* dst = &lds_x[wave][r * 64 + cb];
#pragma unroll
        for (int q = 0; q < 4; ++q) {
            float4 a4 = ph[2 * q + 0], b4 = ph[2 * q + 1];
            float4 c4 = pa[2 * q + 0], d4 = pa[2 * q + 1];
            v8h o;
            o[0] = (_Float16)(a4.x + c4.x); o[1] = (_Float16)(a4.y + c4.y);
            o[2] = (_Float16)(a4.z + c4.z); o[3] = (_Float16)(a4.w + c4.w);
            o[4] = (_Float16)(b4.x + d4.x); o[5] = (_Float16)(b4.y + d4.y);
            o[6] = (_Float16)(b4.z + d4.z); o[7] = (_Float16)(b4.w + d4.w);
            *(v8h*)(dst + q * 8) = o;
        }
    }
    __syncthreads();

    // Stage 1: t = elu( (v@W1 + b1)*scale + be )
    v16h a0 = load_a_frag(lds_x[wave], lane, 0);
    v16h a1 = load_a_frag(lds_x[wave], lane, 32);

    float s1[4][8];
#pragma unroll
    for (int nt = 0; nt < 4; ++nt) {
        v8f c = {};
        v16h bk0 = *(const v16h*)(lds_w1 + (0 * 4 + nt) * 512 + lane * 16);
        c = WMMA_F16(a0, bk0, c);
        v16h bk1 = *(const v16h*)(lds_w1 + (1 * 4 + nt) * 512 + lane * 16);
        c = WMMA_F16(a1, bk1, c);
        int ncol = nt * 16 + (lane & 15);
        float sc = s_scale[ncol], bb = s_b1[ncol], bt = s_be[ncol];
#pragma unroll
        for (int r2 = 0; r2 < 8; ++r2)
            s1[nt][r2] = elu1((c[r2] + bb) * sc + bt);
    }
    __syncthreads();

    // write intermediate tile (C layout: row = r2 + 8*(lane>=16), col = ncol)
    {
        int mb = (lane < 16) ? 0 : 8;
#pragma unroll
        for (int nt = 0; nt < 4; ++nt) {
            int ncol = nt * 16 + (lane & 15);
#pragma unroll
            for (int r2 = 0; r2 < 8; ++r2)
                lds_x[wave][(mb + r2) * 64 + ncol] = (_Float16)s1[nt][r2];
        }
    }
    __syncthreads();

    // Stage 2: hout = elu( t@W2 + b2 )
    v16h u0 = load_a_frag(lds_x[wave], lane, 0);
    v16h u1 = load_a_frag(lds_x[wave], lane, 32);
#pragma unroll
    for (int nt = 0; nt < 4; ++nt) {
        v8f c = {};
        v16h bk0 = *(const v16h*)(lds_w2 + (0 * 4 + nt) * 512 + lane * 16);
        c = WMMA_F16(u0, bk0, c);
        v16h bk1 = *(const v16h*)(lds_w2 + (1 * 4 + nt) * 512 + lane * 16);
        c = WMMA_F16(u1, bk1, c);
        int ncol = nt * 16 + (lane & 15);
        float bb = s_b2[ncol];
        int mb = (lane < 16) ? 0 : 8;
#pragma unroll
        for (int r2 = 0; r2 < 8; ++r2) {
            int node = base + mb + r2;
            float v = elu1(c[r2] + bb);
            if (node < n) hout[(size_t)node * 64 + ncol] = v;
        }
    }
}

// ---------------------------------------------------------------------------
// Pool: segment-sum h over batch + counts (atomics; G=128 rows stay in L2)
// ---------------------------------------------------------------------------
__global__ void pool_kernel(const float* __restrict__ h, const int* __restrict__ batch,
                            float* __restrict__ pooled, float* __restrict__ cnt, int n) {
    int idx = blockIdx.x * 256 + threadIdx.x;
    int node = idx >> 4;
    if (node >= n) return;
    int c = (idx & 15) * 4;
    int gid = batch[node];
    float4 v = *(const float4*)(h + (size_t)node * 64 + c);
    float* p = pooled + (size_t)gid * 64 + c;
    atomicAdd(p + 0, v.x);
    atomicAdd(p + 1, v.y);
    atomicAdd(p + 2, v.z);
    atomicAdd(p + 3, v.w);
    if ((idx & 15) == 0) atomicAdd(cnt + gid, 1.0f);
}

// ---------------------------------------------------------------------------
// Readout: out[g] = ( elu(mean_g @ Wr1 + br1) @ Wr2 + br2 ) @ Wo + bo
// One 64-thread block per graph; tiny, latency-bound.
// ---------------------------------------------------------------------------
__global__ __launch_bounds__(64) void readout(
    const float* __restrict__ pooled, const float* __restrict__ cnt,
    const float* __restrict__ Wr1, const float* __restrict__ br1,
    const float* __restrict__ Wr2, const float* __restrict__ br2,
    const float* __restrict__ Wo,  const float* __restrict__ bo,
    float* __restrict__ out) {
    __shared__ float p[64], r1[64], r2[64];
    int gb = blockIdx.x, j = threadIdx.x;
    float cn = cnt[gb];
    cn = cn > 1.0f ? cn : 1.0f;
    p[j] = pooled[(size_t)gb * 64 + j] / cn;
    __syncthreads();
    float acc = br1[j];
    for (int k = 0; k < 64; ++k) acc += p[k] * Wr1[k * 64 + j];
    r1[j] = elu1(acc);
    __syncthreads();
    float acc2 = br2[j];
    for (int k = 0; k < 64; ++k) acc2 += r1[k] * Wr2[k * 64 + j];
    r2[j] = acc2 * Wo[j];
    __syncthreads();
    for (int s = 32; s > 0; s >>= 1) {
        if (j < s) r2[j] += r2[j + s];
        __syncthreads();
    }
    if (j == 0) out[gb] = r2[0] + bo[0];
}

// ---------------------------------------------------------------------------
extern "C" void kernel_launch(void* const* d_in, const int* in_sizes, int n_in,
                              void* d_out, int out_size, void* d_ws, size_t ws_size,
                              hipStream_t stream) {
    const int* x    = (const int*)d_in[0];
    const int* ei   = (const int*)d_in[1];
    const int* bat  = (const int*)d_in[2];
    const float* emb = (const float*)d_in[3];

    const float* L[3][6];
    for (int l = 0; l < 3; ++l)
        for (int k = 0; k < 6; ++k)
            L[l][k] = (const float*)d_in[4 + l * 6 + k];   // W1,b1,g,be,W2,b2

    const float* Wr1 = (const float*)d_in[22];
    const float* br1 = (const float*)d_in[23];
    const float* Wr2 = (const float*)d_in[24];
    const float* br2 = (const float*)d_in[25];
    const float* Wo  = (const float*)d_in[26];
    const float* bo  = (const float*)d_in[27];
    float* out = (float*)d_out;

    const int N = in_sizes[0];
    const int E = in_sizes[1] / 2;
    const int G = out_size;                       // (G,1) flat
    const size_t N64 = (size_t)N * 64;

    float* hA     = (float*)d_ws;
    float* hB     = hA + N64;
    float* agg    = hB + N64;
    float* pooled = agg + N64;
    float* cnt    = pooled + (size_t)G * 64;

    const int* src = ei;
    const int* dst = ei + E;

    long nThrF = (long)N * 16;
    dim3 blk(256);

    gather_embed<<<dim3((unsigned)((nThrF + 255) / 256)), blk, 0, stream>>>(x, emb, hA, N);

    float* hcur = hA;
    float* hnext = hB;
    long nAgg4 = (long)N * 16;                    // N*64 floats = N*16 float4
    long nEdgeT = (long)E * 16;
    for (int l = 0; l < 3; ++l) {
        zero_f4<<<dim3((unsigned)((nAgg4 + 255) / 256)), blk, 0, stream>>>((float4*)agg, nAgg4);
        scatter_edges<<<dim3((unsigned)((nEdgeT + 255) / 256)), blk, 0, stream>>>(
            hcur, src, dst, agg, E);
        gin_mlp<<<dim3((unsigned)((N + 63) / 64)), dim3(128), 0, stream>>>(
            hcur, agg, L[l][0], L[l][1], L[l][2], L[l][3], L[l][4], L[l][5], hnext, N);
        float* t = hcur; hcur = hnext; hnext = t;
    }

    long nPool = (long)G * 64 + G;
    zero_f1<<<dim3((unsigned)((nPool + 255) / 256)), blk, 0, stream>>>(pooled, nPool);
    pool_kernel<<<dim3((unsigned)((nThrF + 255) / 256)), blk, 0, stream>>>(hcur, bat, pooled, cnt, N);
    readout<<<dim3((unsigned)G), dim3(64), 0, stream>>>(
        pooled, cnt, Wr1, br1, Wr2, br2, Wo, bo, out);
}
